// CausalFlowModel_40003325395644
// MI455X (gfx1250) — compile-verified, matrix operand order checked by reading, not run on hardware
//
#include <hip/hip_runtime.h>
#include <hip/hip_bf16.h>

// Sizes from the reference
#define BB 4096      // batch
#define TT 128       // time steps
#define HH 512       // hidden
#define H4 2048      // 4*H
#define CD1 33       // 1 + CD
#define SD 64        // state dim

typedef __bf16 bf16_t;
typedef __attribute__((ext_vector_type(16))) __bf16 v16bf;
typedef __attribute__((ext_vector_type(8)))  __bf16 v8bf;
typedef __attribute__((ext_vector_type(8)))  float  v8f;
typedef __attribute__((ext_vector_type(4)))  int    v4i;

// CDNA5 async global->LDS copy (ASYNCcnt-tracked, no VGPR round-trip).
// ROCm 7.2 clang-22 prototype (from diagnostics): (v4i AS1*, v4i AS3*, Ii, Ii)
#if defined(__has_builtin)
#  if __has_builtin(__builtin_amdgcn_global_load_async_to_lds_b128)
#    define USE_ASYNC_LDS 1
#  endif
#endif
#ifndef USE_ASYNC_LDS
#  define USE_ASYNC_LDS 0
#endif

#if USE_ASYNC_LDS
typedef __attribute__((address_space(1))) v4i* gptr_v4i;  // global int4*
typedef __attribute__((address_space(3))) v4i* lptr_v4i;  // LDS int4*
#endif

// ---------------------------------------------------------------------------
// fp32 -> bf16 conversion (grid-stride)
// ---------------------------------------------------------------------------
__global__ void cvt_f32_bf16_kernel(const float* __restrict__ in,
                                    bf16_t* __restrict__ out, size_t n) {
  size_t i = (size_t)blockIdx.x * blockDim.x + threadIdx.x;
  size_t stride = (size_t)gridDim.x * blockDim.x;
  for (; i < n; i += stride) out[i] = (bf16_t)in[i];
}

__global__ void zero_f32_kernel(float* __restrict__ p, size_t n) {
  size_t i = (size_t)blockIdx.x * blockDim.x + threadIdx.x;
  size_t stride = (size_t)gridDim.x * blockDim.x;
  for (; i < n; i += stride) p[i] = 0.0f;
}

// ---------------------------------------------------------------------------
// NT GEMM via WMMA bf16: C[m,n] = sum_k A[m,k] * Bw[n,k]  (+ bias, + act)
// A: M x K row-major bf16, Bw: N x K row-major bf16 (i.e. A @ Bw^T).
// BM=128, BN_T in {128,64}, BK=32; 256 threads = 8 waves in 4(m) x 2(n) grid.
// Each wave owns 32 rows x (BN_T/2) cols = 2 x (BN_T/32) WMMA tiles.
// Double-buffered LDS, filled with GLOBAL_LOAD_ASYNC_TO_LDS_B128 when the
// toolchain exposes the builtin (fallback: batched reg-staged loads).
// ACT: 0=none, 1=tanh.  OUT_MODE: 0=f32, 1=bf16, 2=both.
// ---------------------------------------------------------------------------
#define BM 128
#define BK 32

template <int ACT, int OUT_MODE, int BN_T>
__global__ __launch_bounds__(256)
void gemm_nt_bf16_kernel(const bf16_t* __restrict__ A,
                         const bf16_t* __restrict__ Bw,
                         const float* __restrict__ bias,
                         float* __restrict__ Cf,
                         bf16_t* __restrict__ Cb,
                         int M, int N, int K) {
  constexpr int NT  = BN_T / 32;   // per-wave N tiles (2 wave cols span BN_T)
  constexpr int ACH = 2;           // A 16B-chunks per thread: 128*32*2/16/256
  constexpr int BCH = BN_T / 64;   // B 16B-chunks per thread

  __shared__ bf16_t sA[2][BM][BK];
  __shared__ bf16_t sB[2][BN_T][BK];

  const int tid  = threadIdx.x;
  const int lane = tid & 31;
  const int wave = tid >> 5;      // 0..7
  const int wm   = wave & 3;      // 0..3 : 32-row band within BM
  const int wn   = wave >> 2;     // 0..1 : (NT*16)-col band within BN_T
  const int block_m = blockIdx.y * BM;
  const int block_n = blockIdx.x * BN_T;

  const int lrow = lane & 15;             // 0..15
  const int k8   = (lane < 16) ? 0 : 8;   // K base per ISA A-frag layout

  v8f acc[2][NT];
#pragma unroll
  for (int i = 0; i < 2; ++i)
#pragma unroll
    for (int j = 0; j < NT; ++j)
#pragma unroll
      for (int r = 0; r < 8; ++r) acc[i][j][r] = 0.0f;

  // ---- stage one K-tile (BM x BK of A, BN_T x BK of B) into LDS buf ----
  auto stage = [&](int kk, int buf) {
#if USE_ASYNC_LDS
#pragma unroll
    for (int i = 0; i < ACH; ++i) {
      int c = tid + i * 256, row = c >> 2, col8 = (c & 3) << 3;
      __builtin_amdgcn_global_load_async_to_lds_b128(
          (gptr_v4i)(A + (size_t)(block_m + row) * K + kk + col8),
          (lptr_v4i)(&sA[buf][row][col8]), 0, 0);
    }
#pragma unroll
    for (int i = 0; i < BCH; ++i) {
      int c = tid + i * 256, row = c >> 2, col8 = (c & 3) << 3;
      __builtin_amdgcn_global_load_async_to_lds_b128(
          (gptr_v4i)(Bw + (size_t)(block_n + row) * K + kk + col8),
          (lptr_v4i)(&sB[buf][row][col8]), 0, 0);
    }
#else
    // fallback: batch all global loads into distinct regs, then all ds writes
    v8bf ra[ACH], rb[BCH];
#pragma unroll
    for (int i = 0; i < ACH; ++i) {
      int c = tid + i * 256, row = c >> 2, col8 = (c & 3) << 3;
      ra[i] = *(const v8bf*)&A[(size_t)(block_m + row) * K + kk + col8];
    }
#pragma unroll
    for (int i = 0; i < BCH; ++i) {
      int c = tid + i * 256, row = c >> 2, col8 = (c & 3) << 3;
      rb[i] = *(const v8bf*)&Bw[(size_t)(block_n + row) * K + kk + col8];
    }
#pragma unroll
    for (int i = 0; i < ACH; ++i) {
      int c = tid + i * 256, row = c >> 2, col8 = (c & 3) << 3;
      *(v8bf*)&sA[buf][row][col8] = ra[i];
    }
#pragma unroll
    for (int i = 0; i < BCH; ++i) {
      int c = tid + i * 256, row = c >> 2, col8 = (c & 3) << 3;
      *(v8bf*)&sB[buf][row][col8] = rb[i];
    }
#endif
  };

  const int nIter = K / BK;
  stage(0, 0);
#if USE_ASYNC_LDS
  asm volatile("s_wait_asynccnt 0" ::: "memory");
#endif
  __syncthreads();

  for (int it = 0; it < nIter; ++it) {
    const int cur = it & 1;
    if (it + 1 < nIter) stage((it + 1) * BK, cur ^ 1);  // overlap with compute

    // ---- build fragments from LDS and issue WMMAs ----
    v16bf afrag[2], bfrag[NT];
#pragma unroll
    for (int i = 0; i < 2; ++i) {
      int r = wm * 32 + i * 16 + lrow;
      v8bf lo = *(const v8bf*)&sA[cur][r][k8];
      v8bf hi = *(const v8bf*)&sA[cur][r][k8 + 16];
      afrag[i] = __builtin_shufflevector(lo, hi, 0, 1, 2, 3, 4, 5, 6, 7,
                                         8, 9, 10, 11, 12, 13, 14, 15);
    }
#pragma unroll
    for (int j = 0; j < NT; ++j) {
      int ccol = wn * (NT * 16) + j * 16 + lrow;
      v8bf lo = *(const v8bf*)&sB[cur][ccol][k8];
      v8bf hi = *(const v8bf*)&sB[cur][ccol][k8 + 16];
      bfrag[j] = __builtin_shufflevector(lo, hi, 0, 1, 2, 3, 4, 5, 6, 7,
                                         8, 9, 10, 11, 12, 13, 14, 15);
    }
#pragma unroll
    for (int i = 0; i < 2; ++i)
#pragma unroll
      for (int j = 0; j < NT; ++j)
        acc[i][j] = __builtin_amdgcn_wmma_f32_16x16x32_bf16(
            false, afrag[i], false, bfrag[j], (short)0, acc[i][j], false, false);

#if USE_ASYNC_LDS
    asm volatile("s_wait_asynccnt 0" ::: "memory");
#endif
    __syncthreads();
  }

  // ---- epilogue: bias + activation + store (f32 / bf16 / both) ----
  const int mhalf = (lane < 16) ? 0 : 8;
#pragma unroll
  for (int i = 0; i < 2; ++i) {
#pragma unroll
    for (int j = 0; j < NT; ++j) {
      const int col = block_n + wn * (NT * 16) + j * 16 + lrow;
      const float bv = bias ? bias[col] : 0.0f;
#pragma unroll
      for (int r = 0; r < 8; ++r) {
        const int row = block_m + wm * 32 + i * 16 + mhalf + r;
        float v = acc[i][j][r] + bv;
        if (ACT == 1) v = tanhf(v);
        const size_t idx = (size_t)row * N + col;
        if (OUT_MODE == 0) {
          Cf[idx] = v;
        } else if (OUT_MODE == 1) {
          Cb[idx] = (bf16_t)v;
        } else {
          Cf[idx] = v;
          Cb[idx] = (bf16_t)v;
        }
      }
    }
  }
}

// ---------------------------------------------------------------------------
// LSTM-ish gate kernel: one block per batch row b, 256 threads, 2 j's each.
// preact holds h@Wh^T (f32).  Adds bi+bh and the K=33 u@Wi^T contribution
// (u row cached in LDS), applies gates, updates c and h, writes bf16 h for
// the next step's WMMA GEMM.
// ---------------------------------------------------------------------------
__global__ __launch_bounds__(256)
void lstm_gates_kernel(const float* __restrict__ preact,  // B x 4H
                       const float* __restrict__ u,       // B x 33 (step slice)
                       const float* __restrict__ Wi,      // 4H x 33
                       const float* __restrict__ bi,
                       const float* __restrict__ bh,
                       float* __restrict__ h,             // B x H (f32 carry)
                       float* __restrict__ c,             // B x H
                       bf16_t* __restrict__ h_bf)         // B x H (bf16 mirror)
{
  __shared__ float su[CD1];
  const int b   = blockIdx.x;
  const int tid = threadIdx.x;
  if (tid < CD1) su[tid] = u[(size_t)b * CD1 + tid];
  __syncthreads();
  const float d = su[CD1 - 1];  // d = u[:, -1]

#pragma unroll
  for (int jj = 0; jj < 2; ++jj) {
    const int j = tid + jj * 256;  // 0..511
    float z[4];
#pragma unroll
    for (int g = 0; g < 4; ++g) {
      const int o = g * HH + j;
      float s = preact[(size_t)b * H4 + o] + bi[o] + bh[o];
      const float* wr = &Wi[(size_t)o * CD1];
#pragma unroll
      for (int k = 0; k < CD1; ++k) s += su[k] * wr[k];
      z[g] = s;
    }
    const float it = 1.0f / (1.0f + __expf(-z[0]));
    const float ft = 1.0f / (1.0f + __expf(-z[1]));
    const float ot = 1.0f / (1.0f + __expf(-z[2]));
    const float gt = tanhf(z[3]);
    const size_t idx = (size_t)b * HH + j;
    const float cn = c[idx] * ft + it * gt;
    const float hn = h[idx] + d * (ot * tanhf(cn));
    c[idx]    = cn;
    h[idx]    = hn;
    h_bf[idx] = (bf16_t)hn;
  }
}

// ---------------------------------------------------------------------------
// Host-side orchestration
// ---------------------------------------------------------------------------
static inline void launch_cvt(const float* in, bf16_t* out, size_t n,
                              hipStream_t stream) {
  cvt_f32_bf16_kernel<<<1024, 256, 0, stream>>>(in, out, n);
}

extern "C" void kernel_launch(void* const* d_in, const int* in_sizes, int n_in,
                              void* d_out, int out_size, void* d_ws, size_t ws_size,
                              hipStream_t stream) {
  // Inputs in setup_inputs() dict order (lists flattened):
  const float* x    = (const float*)d_in[0];   // (B, 64)
  const float* rnn  = (const float*)d_in[1];   // (T, B, 33)
  // d_in[2] = deltas, unused by the reference step()
  const float* Wi   = (const float*)d_in[3];   // (2048, 33)
  const float* bi   = (const float*)d_in[4];   // (2048,)
  const float* Wh   = (const float*)d_in[5];   // (2048, 512)
  const float* bh   = (const float*)d_in[6];   // (2048,)
  const float* eW0  = (const float*)d_in[7];   // (2048, 64)
  const float* eW1  = (const float*)d_in[8];   // (2048, 2048)
  const float* eW2  = (const float*)d_in[9];   // (512, 2048)
  const float* eb0  = (const float*)d_in[10];
  const float* eb1  = (const float*)d_in[11];
  const float* eb2  = (const float*)d_in[12];
  const float* dW0  = (const float*)d_in[13];  // (2048, 512)
  const float* dW1  = (const float*)d_in[14];  // (2048, 2048)
  const float* dW2  = (const float*)d_in[15];  // (64, 2048)
  const float* db0  = (const float*)d_in[16];
  const float* db1  = (const float*)d_in[17];
  const float* db2  = (const float*)d_in[18];
  float* out = (float*)d_out;                  // (B, 64) f32

  // ---- carve workspace ----
  size_t off = 0;
  char* base = (char*)d_ws;
  auto carve = [&](size_t bytes) {
    void* p = base + off;
    off += (bytes + 255) & ~(size_t)255;
    return p;
  };
  bf16_t* x_bf   = (bf16_t*)carve((size_t)BB * SD * 2);
  bf16_t* eW0_bf = (bf16_t*)carve((size_t)H4 * SD * 2);
  bf16_t* eW1_bf = (bf16_t*)carve((size_t)H4 * H4 * 2);
  bf16_t* eW2_bf = (bf16_t*)carve((size_t)HH * H4 * 2);
  bf16_t* dW0_bf = (bf16_t*)carve((size_t)H4 * HH * 2);
  bf16_t* dW1_bf = (bf16_t*)carve((size_t)H4 * H4 * 2);
  bf16_t* dW2_bf = (bf16_t*)carve((size_t)SD * H4 * 2);
  bf16_t* Wh_bf  = (bf16_t*)carve((size_t)H4 * HH * 2);
  bf16_t* act1   = (bf16_t*)carve((size_t)BB * H4 * 2);
  bf16_t* act2   = (bf16_t*)carve((size_t)BB * H4 * 2);
  float*  preact = (float*)carve((size_t)BB * H4 * 4);
  float*  h_f32  = (float*)carve((size_t)BB * HH * 4);
  float*  c_f32  = (float*)carve((size_t)BB * HH * 4);
  bf16_t* h_bf   = (bf16_t*)carve((size_t)BB * HH * 2);
  (void)ws_size;

  // ---- convert inputs/weights to bf16 ----
  launch_cvt(x,   x_bf,   (size_t)BB * SD, stream);
  launch_cvt(eW0, eW0_bf, (size_t)H4 * SD, stream);
  launch_cvt(eW1, eW1_bf, (size_t)H4 * H4, stream);
  launch_cvt(eW2, eW2_bf, (size_t)HH * H4, stream);
  launch_cvt(dW0, dW0_bf, (size_t)H4 * HH, stream);
  launch_cvt(dW1, dW1_bf, (size_t)H4 * H4, stream);
  launch_cvt(dW2, dW2_bf, (size_t)SD * H4, stream);
  launch_cvt(Wh,  Wh_bf,  (size_t)H4 * HH, stream);
  zero_f32_kernel<<<1024, 256, 0, stream>>>(c_f32, (size_t)BB * HH);

  const dim3 blk(256);
  auto grid_for = [](int M, int N, int bn) {
    return dim3((unsigned)(N / bn), (unsigned)(M / BM));
  };

  // ---- encoder: tanh(x@W0^T+b0) -> tanh(.@W1^T+b1) -> .@W2^T+b2 = h0 ----
  gemm_nt_bf16_kernel<1, 1, 128><<<grid_for(BB, H4, 128), blk, 0, stream>>>(
      x_bf, eW0_bf, eb0, nullptr, act1, BB, H4, SD);
  gemm_nt_bf16_kernel<1, 1, 128><<<grid_for(BB, H4, 128), blk, 0, stream>>>(
      act1, eW1_bf, eb1, nullptr, act2, BB, H4, H4);
  gemm_nt_bf16_kernel<0, 2, 128><<<grid_for(BB, HH, 128), blk, 0, stream>>>(
      act2, eW2_bf, eb2, h_f32, h_bf, BB, HH, H4);

  // ---- recurrent scan: preact = h@Wh^T (WMMA), then gates (VALU) ----
  for (int t = 0; t < TT; ++t) {
    gemm_nt_bf16_kernel<0, 0, 128><<<grid_for(BB, H4, 128), blk, 0, stream>>>(
        h_bf, Wh_bf, nullptr, preact, nullptr, BB, H4, HH);
    const float* u_t = rnn + (size_t)t * BB * CD1;
    lstm_gates_kernel<<<BB, blk, 0, stream>>>(preact, u_t, Wi, bi, bh,
                                              h_f32, c_f32, h_bf);
  }

  // ---- decoder: tanh(h@W0^T+b0) -> tanh(.@W1^T+b1) -> .@W2^T+b2 -> out ----
  gemm_nt_bf16_kernel<1, 1, 128><<<grid_for(BB, H4, 128), blk, 0, stream>>>(
      h_bf, dW0_bf, db0, nullptr, act1, BB, H4, HH);
  gemm_nt_bf16_kernel<1, 1, 128><<<grid_for(BB, H4, 128), blk, 0, stream>>>(
      act1, dW1_bf, db1, nullptr, act2, BB, H4, H4);
  gemm_nt_bf16_kernel<0, 0, 64><<<grid_for(BB, SD, 64), blk, 0, stream>>>(
      act2, dW2_bf, db2, out, nullptr, BB, SD, H4);
}
